// VisionTransformer_55327768708404
// MI455X (gfx1250) — matrix operand order repaired and checked by reading, not compiled, not run on hardware
//
#include <hip/hip_runtime.h>
#include <stdint.h>

// ---------- CDNA5 WMMA types ----------
typedef __attribute__((ext_vector_type(16))) __bf16 v16bf;
typedef __attribute__((ext_vector_type(8)))  float  v8f;

union FragU { uint4 q[2]; v16bf v; };

static __device__ inline v16bf load_frag(const uint16_t* p0, const uint16_t* p1) {
  FragU f;
  f.q[0] = *(const uint4*)p0;
  f.q[1] = *(const uint4*)p1;
  return f.v;
}

#define WMMA_BF16(A, B, C) \
  __builtin_amdgcn_wmma_f32_16x16x32_bf16(false, (A), false, (B), (short)0, (C), false, false)

static __device__ inline uint16_t f2bf(float x) {  // round-to-nearest-even
  uint32_t u = __float_as_uint(x);
  u += 0x7FFFu + ((u >> 16) & 1u);
  return (uint16_t)(u >> 16);
}
static __device__ inline int imin(int a, int b) { return a < b ? a : b; }
static __device__ inline int imax(int a, int b) { return a > b ? a : b; }
static __device__ inline int iabs(int a) { return a < 0 ? -a : a; }

// ---------- CDNA5 async global->LDS (ASYNCcnt-tracked, §15.18.3 op 98) ----------
// Generic pointers to LDS carry the LDS byte offset in bits [31:0] (ISA §10.2).
static __device__ inline uint32_t lds_off32(const void* p) {
  return (uint32_t)(uintptr_t)p;
}
static __device__ inline void async_ld16(uint32_t loff, const void* gaddr) {
  asm volatile("global_load_async_to_lds_b128 %0, %1, off"
               :: "v"(loff), "v"(gaddr) : "memory");
}
static __device__ inline void wait_async0() {
  asm volatile("s_wait_asynccnt 0" ::: "memory");
}

// ---------- problem constants ----------
#define BATCH 8
#define NTOK  1025          // 1 cls + 32*32 grid
#define CDIM  768
#define HEADS 12
#define HD    64
#define ROWS  (BATCH * NTOK)     // 8200
#define NPAD  1056               // keys padded to multiple of 32

// ---------- f32 -> bf16 convert ----------
__global__ void k_f32_to_bf16(const float* __restrict__ in, uint16_t* __restrict__ out, int n) {
  int i = blockIdx.x * blockDim.x + threadIdx.x;
  int stride = gridDim.x * blockDim.x;
  for (; i < n; i += stride) out[i] = f2bf(in[i]);
}

// ---------- bias table: geometry collapses to (|dpx|,|dpy|) in [0,32)^2 ----------
__global__ void k_bias_tab(const float* __restrict__ Wg, const float* __restrict__ bg,
                           float* __restrict__ tab) {
  int gid = blockIdx.x * blockDim.x + threadIdx.x;
  if (gid >= HEADS * 1024) return;
  int h = gid >> 10, rem = gid & 1023;
  int a = rem >> 5, bcol = rem & 31;
  // |cx_i - cx_j| / w = |dpx| / 33 ; dw = dh = 0 (uniform cell sizes)
  float dx = __logf(fmaxf((float)a * (1.0f / 33.0f), 0.001f));
  float dy = __logf(fmaxf((float)bcol * (1.0f / 33.0f), 0.001f));
  const float* wg = Wg + h * 64;
  float acc = bg[h];
#pragma unroll
  for (int d = 48; d < 64; ++d) acc += wg[d];   // cos(0)=1 channels; sin(0)=0 channels drop
#pragma unroll
  for (int r = 0; r < 8; ++r) {
    float dm = __expf(-0.86346936f * (float)r);  // 1000^(-r/8)
    float sx, cx, sy, cy;
    __sincosf(100.0f * dx * dm, &sx, &cx);
    __sincosf(100.0f * dy * dm, &sy, &cy);
    acc += sx * wg[r] + sy * wg[8 + r] + cx * wg[32 + r] + cy * wg[40 + r];
  }
  tab[gid] = __logf(fmaxf(fmaxf(acc, 0.0f), 1e-6f));
}

// ----------------------------------------------------------------------
// Shared GEMM step: compile-time buffer index CUR (static LDS offsets),
// B-fragments software-pipelined one ahead of the WMMA chain.
// ----------------------------------------------------------------------
#define GEMM_STEP(CUR, KK, KMAX)                                              \
  {                                                                           \
    wait_async0();                                                            \
    __syncthreads();                                                          \
    if ((KK) + 32 < (KMAX)) {                                                 \
      const uint16_t* src = bcol + (KK) + 32;                                 \
      uint32_t dst = bco + ((CUR) ^ 1) * (128 * 32 * 2);                      \
      _Pragma("unroll")                                                       \
      for (int q = 0; q < 4; ++q) async_ld16(dst + q * 16, src + q * 8);      \
    }                                                                         \
    v16bf a = load_frag(arow + (KK) + half * 16, arow + (KK) + half * 16 + 8); \
    const uint16_t* bbase = &bst[(CUR)][nlo][half * 16];                      \
    v16bf bf = load_frag(bbase, bbase + 8);                                   \
    _Pragma("unroll")                                                         \
    for (int nt = 0; nt < 8; ++nt) {                                          \
      v16bf bfn = bf;                                                         \
      if (nt < 7) {                                                           \
        const uint16_t* bp = bbase + (nt + 1) * (16 * 32);                    \
        bfn = load_frag(bp, bp + 8);                                          \
      }                                                                       \
      acc[nt] = WMMA_BF16(a, bf, acc[nt]);                                    \
      bf = bfn;                                                               \
    }                                                                         \
  }

// ======================================================================
// QKV projection: (8200x768) x (768x2304). 4-wave block computes 64x128;
// the 128x32 weight tile is async-staged into LDS (double buffered).
// ======================================================================
__global__ __launch_bounds__(128) void k_qkv(const uint16_t* __restrict__ xb,
                                             const uint16_t* __restrict__ wb,
                                             uint16_t* __restrict__ Qb,
                                             uint16_t* __restrict__ Kb,
                                             uint16_t* __restrict__ Vt) {
  __shared__ __align__(16) uint16_t bst[2][128][32];   // [buf][col][k] : 16 KB
  const int tid  = threadIdx.x;
  const int wid  = tid >> 5, lane = tid & 31;
  const int nlo  = lane & 15, half = lane >> 4;
  const int rowt = blockIdx.x * 4 + wid, colg = blockIdx.y;
  const int rowA = imin(rowt * 16 + nlo, ROWS - 1);
  const uint16_t* arow = xb + (size_t)rowA * CDIM;
  const uint16_t* bcol = wb + (size_t)(colg * 128 + tid) * CDIM;  // this thread's staged column
  uint32_t bco = lds_off32(&bst[0][tid][0]);

  v8f acc[8];
#pragma unroll
  for (int i = 0; i < 8; ++i) acc[i] = {};

  // prologue: stage k-block 0 into buffer 0 (4 x b128 per thread = 8 KB total)
#pragma unroll
  for (int q = 0; q < 4; ++q) async_ld16(bco + q * 16, bcol + q * 8);

  for (int it2 = 0; it2 < 12; ++it2) {   // 24 k-steps, buffer parity static
    const int kk = it2 * 64;
    GEMM_STEP(0, kk, CDIM);
    GEMM_STEP(1, kk + 32, CDIM);
  }
  // scatter into Q (scaled), K row-major, V transposed
#pragma unroll
  for (int nt = 0; nt < 8; ++nt) {
    int o = colg * 128 + nt * 16 + nlo;
    int s = o / CDIM, hh = (o % CDIM) >> 6, d = o & 63;
    float mul = (s == 0) ? 0.125f : 1.0f;  // hd^-0.5 folded into Q
#pragma unroll
    for (int r = 0; r < 8; ++r) {
      int row = rowt * 16 + r + half * 8;
      if (row < ROWS) {
        int b = row / NTOK, n = row % NTOK;
        uint16_t val = f2bf(acc[nt][r] * mul);
        size_t bh = (size_t)(b * HEADS + hh);
        if (s == 0)      Qb[(bh * NTOK + n) * HD + d] = val;
        else if (s == 1) Kb[(bh * NPAD + n) * HD + d] = val;
        else             Vt[(bh * HD + d) * NPAD + n] = val;
      }
    }
  }
}

// ----------------------------------------------------------------------
// One attention chunk with compile-time buffer index CUR.
// All 4 K fragments loaded before the 4 logits WMMAs (s0/s1 interleaved);
// all 4 V fragments loaded before the 4 PV WMMAs (independent oacc).
// ----------------------------------------------------------------------
#define ATTN_CHUNK(M0, CUR)                                                   \
  {                                                                           \
    wait_async0();                                                            \
    __syncthreads();                                                          \
    if ((M0) + 32 <= 1024) {                                                  \
      const int m1 = (M0) + 32;                                               \
      const uint32_t kon = ko0 + ((CUR) ^ 1) * (32 * 64 * 2);                 \
      const uint32_t von = vo0 + ((CUR) ^ 1) * (64 * 32 * 2);                 \
      const uint16_t* kr = kbase + (size_t)(m1 + kj) * HD;                    \
      async_ld16(kon + kq * 16,       kr + kq * 8);                           \
      async_ld16(kon + (kq + 4) * 16, kr + (kq + 4) * 8);                     \
      const uint16_t* vr = vbase + (size_t)vd * NPAD + m1;                    \
      async_ld16(von + vc * 16,       vr + vc * 8);                           \
      async_ld16(von + (vc + 2) * 16, vr + (vc + 2) * 8);                     \
    }                                                                         \
    const uint16_t* kpA = &kst[(CUR)][nlo][half * 16];                        \
    const uint16_t* kpB = &kst[(CUR)][16 + nlo][half * 16];                   \
    v16bf k00 = load_frag(kpA, kpA + 8);                                      \
    v16bf k01 = load_frag(kpA + 32, kpA + 40);                                \
    v16bf k10 = load_frag(kpB, kpB + 8);                                      \
    v16bf k11 = load_frag(kpB + 32, kpB + 40);                                \
    v8f s0 = {}, s1 = {};                                                     \
    s0 = WMMA_BF16(qa0, k00, s0);                                             \
    s1 = WMMA_BF16(qa0, k10, s1);                                             \
    s0 = WMMA_BF16(qa1, k01, s0);                                             \
    s1 = WMMA_BF16(qa1, k11, s1);                                             \
    _Pragma("unroll")                                                         \
    for (int t = 0; t < 2; ++t) {                                             \
      v8f& st = (t == 0) ? s0 : s1;                                           \
      int mt = (M0) + t * 16 + nlo;                                           \
      if (mt >= NTOK) {                                                       \
        _Pragma("unroll")                                                     \
        for (int r = 0; r < 8; ++r) st[r] = -1e30f;                           \
      } else if (mt > 0) {                                                    \
        int mi = mt - 1, pxm = mi >> 5, pym = mi & 31;                        \
        _Pragma("unroll")                                                     \
        for (int r = 0; r < 8; ++r)                                           \
          st[r] += nz[r] ? 0.0f                                               \
                         : tabs[iabs(pxn[r] - pxm) * 32 + iabs(pyn[r] - pym)];\
      }                                                                       \
    }                                                                         \
    float cmax[8];                                                            \
    _Pragma("unroll")                                                         \
    for (int r = 0; r < 8; ++r) cmax[r] = fmaxf(s0[r], s1[r]);                \
    _Pragma("unroll")                                                         \
    for (int mk = 1; mk <= 8; mk <<= 1)                                       \
      _Pragma("unroll")                                                       \
      for (int r = 0; r < 8; ++r)                                             \
        cmax[r] = fmaxf(cmax[r], __shfl_xor(cmax[r], mk, 32));                \
    float al[8];                                                              \
    _Pragma("unroll")                                                         \
    for (int r = 0; r < 8; ++r) {                                             \
      float mn = fmaxf(m_r[r], cmax[r]);                                      \
      al[r] = __expf(m_r[r] - mn);                                            \
      m_r[r] = mn;                                                            \
    }                                                                         \
    float rs[8];                                                              \
    _Pragma("unroll")                                                         \
    for (int r = 0; r < 8; ++r) {                                             \
      s0[r] = __expf(s0[r] - m_r[r]);                                         \
      s1[r] = __expf(s1[r] - m_r[r]);                                         \
      rs[r] = s0[r] + s1[r];                                                  \
    }                                                                         \
    _Pragma("unroll")                                                         \
    for (int mk = 1; mk <= 8; mk <<= 1)                                       \
      _Pragma("unroll")                                                       \
      for (int r = 0; r < 8; ++r) rs[r] += __shfl_xor(rs[r], mk, 32);         \
    _Pragma("unroll")                                                         \
    for (int r = 0; r < 8; ++r) l_r[r] = l_r[r] * al[r] + rs[r];              \
    _Pragma("unroll")                                                         \
    for (int dt = 0; dt < 4; ++dt)                                            \
      _Pragma("unroll")                                                       \
      for (int r = 0; r < 8; ++r) oacc[dt][r] *= al[r];                       \
    _Pragma("unroll")                                                         \
    for (int r = 0; r < 8; ++r) {                                             \
      pst[wid][r + half * 8][nlo]      = f2bf(s0[r]);                         \
      pst[wid][r + half * 8][16 + nlo] = f2bf(s1[r]);                         \
    }                                                                         \
    asm volatile("" ::: "memory");                                            \
    const uint16_t* pp = &pst[wid][nlo][0];                                   \
    v16bf pa = load_frag(pp + half * 16, pp + half * 16 + 8);                 \
    const uint16_t* vp0 = &vst[(CUR)][nlo][half * 16];                        \
    v16bf vf0 = load_frag(vp0,                 vp0 + 8);                      \
    v16bf vf1 = load_frag(vp0 + 16 * 32,       vp0 + 16 * 32 + 8);            \
    v16bf vf2 = load_frag(vp0 + 32 * 32,       vp0 + 32 * 32 + 8);            \
    v16bf vf3 = load_frag(vp0 + 48 * 32,       vp0 + 48 * 32 + 8);            \
    oacc[0] = WMMA_BF16(pa, vf0, oacc[0]);                                    \
    oacc[1] = WMMA_BF16(pa, vf1, oacc[1]);                                    \
    oacc[2] = WMMA_BF16(pa, vf2, oacc[2]);                                    \
    oacc[3] = WMMA_BF16(pa, vf3, oacc[3]);                                    \
  }

// ======================================================================
// Flash attention. 4 waves/block share (b,h): the 32-key K (32x64) and
// V (64x32) chunks are async-staged into LDS (double buffered) once per
// block and consumed by all 4 waves via ds_load_b128 fragments.
// ======================================================================
__global__ __launch_bounds__(128) void k_attn(const uint16_t* __restrict__ Qb,
                                              const uint16_t* __restrict__ Kb,
                                              const uint16_t* __restrict__ Vt,
                                              const float* __restrict__ tab,
                                              uint16_t* __restrict__ Ob) {
  __shared__ float tabs[1024];
  __shared__ __align__(16) uint16_t pst[4][16][32];    // per-wave P bounce
  __shared__ __align__(16) uint16_t kst[2][32][64];    // [buf][key][d]  : 8 KB
  __shared__ __align__(16) uint16_t vst[2][64][32];    // [buf][d][key]  : 8 KB
  const int bh = blockIdx.x, b = bh / HEADS, h = bh % HEADS;
  const int tid = threadIdx.x;
  for (int i = tid; i < 1024; i += 128) tabs[i] = tab[h * 1024 + i];
  const int wid = tid >> 5, lane = tid & 31;
  const int nlo = lane & 15, half = lane >> 4;
  const int qt = imin(blockIdx.y * 4 + wid, 64);  // clamp (no early return: barriers)

  const size_t bhs = (size_t)bh;
  const uint16_t* kbase = Kb + bhs * NPAD * HD;
  const uint16_t* vbase = Vt + bhs * HD * NPAD;

  // per-thread async staging assignments (whole block covers one chunk)
  const int kj = tid >> 2, kq = tid & 3;        // K: row kj (0..31), 16B chunks kq,kq+4
  const int vd = tid >> 1, vc = tid & 1;        // V: row vd (0..63), 16B chunks vc,vc+2
  const uint32_t ko0 = lds_off32(&kst[0][kj][0]);
  const uint32_t vo0 = lds_off32(&vst[0][vd][0]);

  // Q fragments (A-layout): rows indexed by lane&15, clamped at edge
  const uint16_t* qrow = Qb + (bhs * NTOK + imin(qt * 16 + nlo, NTOK - 1)) * HD;
  v16bf qa0 = load_frag(qrow + half * 16,      qrow + half * 16 + 8);
  v16bf qa1 = load_frag(qrow + 32 + half * 16, qrow + 40 + half * 16);

  // C-layout row geometry for bias lookups
  int qn[8], pxn[8], pyn[8]; bool nz[8];
#pragma unroll
  for (int r = 0; r < 8; ++r) {
    qn[r] = qt * 16 + r + half * 8;
    int nc = imin(qn[r], NTOK - 1);
    nz[r] = (nc == 0);
    int i = imax(nc - 1, 0);
    pxn[r] = i >> 5; pyn[r] = i & 31;
  }

  float m_r[8], l_r[8]; v8f oacc[4];
#pragma unroll
  for (int r = 0; r < 8; ++r) { m_r[r] = -1e30f; l_r[r] = 0.0f; }
#pragma unroll
  for (int dt = 0; dt < 4; ++dt) oacc[dt] = {};

  // prologue: stage chunk 0 into buffer 0
  {
    const uint16_t* kr = kbase + (size_t)kj * HD;
    async_ld16(ko0 + kq * 16,        kr + kq * 8);
    async_ld16(ko0 + (kq + 4) * 16,  kr + (kq + 4) * 8);
    const uint16_t* vr = vbase + (size_t)vd * NPAD;
    async_ld16(vo0 + vc * 16,        vr + vc * 8);
    async_ld16(vo0 + (vc + 2) * 16,  vr + (vc + 2) * 8);
  }

  for (int p = 0; p < 16; ++p) {       // 32 chunks, buffer parity static
    const int m0 = p * 64;
    ATTN_CHUNK(m0, 0);
    ATTN_CHUNK(m0 + 32, 1);
  }
  ATTN_CHUNK(1024, 0);                 // tail chunk (key 1024 + mask)

  // normalize + store bf16 (row-major O for the projection GEMM).
  // Clamped edge waves write bitwise-identical duplicates for row 1024.
  const int col0 = h * HD;
#pragma unroll
  for (int r = 0; r < 8; ++r) {
    float inv = 1.0f / fmaxf(l_r[r], 1e-30f);
    if (qn[r] < NTOK) {
      size_t row = (size_t)b * NTOK + qn[r];
#pragma unroll
      for (int dt = 0; dt < 4; ++dt)
        Ob[row * CDIM + col0 + dt * 16 + nlo] = f2bf(oacc[dt][r] * inv);
    }
  }
}

// ======================================================================
// Output projection: (8200x768) x (768x768) + bproj, async-staged B tile
// ======================================================================
__global__ __launch_bounds__(128) void k_proj(const uint16_t* __restrict__ Ob,
                                              const uint16_t* __restrict__ wb,
                                              const float* __restrict__ bproj,
                                              float* __restrict__ out) {
  __shared__ __align__(16) uint16_t bst[2][128][32];
  const int tid  = threadIdx.x;
  const int wid  = tid >> 5, lane = tid & 31;
  const int nlo  = lane & 15, half = lane >> 4;
  const int rowt = blockIdx.x * 4 + wid, colg = blockIdx.y;
  const int rowA = imin(rowt * 16 + nlo, ROWS - 1);
  const uint16_t* arow = Ob + (size_t)rowA * CDIM;
  const uint16_t* bcol = wb + (size_t)(colg * 128 + tid) * CDIM;
  uint32_t bco = lds_off32(&bst[0][tid][0]);

  v8f acc[8];
#pragma unroll
  for (int i = 0; i < 8; ++i) acc[i] = {};

#pragma unroll
  for (int q = 0; q < 4; ++q) async_ld16(bco + q * 16, bcol + q * 8);

  for (int it2 = 0; it2 < 12; ++it2) {
    const int kk = it2 * 64;
    GEMM_STEP(0, kk, CDIM);
    GEMM_STEP(1, kk + 32, CDIM);
  }
#pragma unroll
  for (int nt = 0; nt < 8; ++nt) {
    int col = colg * 128 + nt * 16 + nlo;
    float bp = bproj[col];
#pragma unroll
    for (int r = 0; r < 8; ++r) {
      int row = rowt * 16 + r + half * 8;
      if (row < ROWS) out[(size_t)row * CDIM + col] = acc[nt][r] + bp;
    }
  }
}

// ---------- workspace layout (bytes, 256-aligned) ----------
#define OFF_TAB   0u
#define OFF_XB    49152u
#define OFF_WQKV  12644352u
#define OFF_WPROJ 16183296u
#define OFF_QB    17362944u
#define OFF_KB    29958144u
#define OFF_VT    42934272u
#define OFF_OB    55910400u
// total = 68,505,600 bytes

extern "C" void kernel_launch(void* const* d_in, const int* in_sizes, int n_in,
                              void* d_out, int out_size, void* d_ws, size_t ws_size,
                              hipStream_t stream) {
  const float* x     = (const float*)d_in[0];
  const float* Wqkv  = (const float*)d_in[1];
  const float* Wproj = (const float*)d_in[2];
  const float* bproj = (const float*)d_in[3];
  const float* Wg    = (const float*)d_in[4];
  const float* bg    = (const float*)d_in[5];
  float* out = (float*)d_out;

  char* w = (char*)d_ws;
  float*    tab    = (float*)(w + OFF_TAB);
  uint16_t* xb     = (uint16_t*)(w + OFF_XB);
  uint16_t* wqkvb  = (uint16_t*)(w + OFF_WQKV);
  uint16_t* wprojb = (uint16_t*)(w + OFF_WPROJ);
  uint16_t* Qb     = (uint16_t*)(w + OFF_QB);
  uint16_t* Kb     = (uint16_t*)(w + OFF_KB);
  uint16_t* Vt     = (uint16_t*)(w + OFF_VT);
  uint16_t* Ob     = (uint16_t*)(w + OFF_OB);

  k_f32_to_bf16<<<4096, 256, 0, stream>>>(x, xb, ROWS * CDIM);
  k_f32_to_bf16<<<2048, 256, 0, stream>>>(Wqkv, wqkvb, 3 * CDIM * CDIM);
  k_f32_to_bf16<<<1024, 256, 0, stream>>>(Wproj, wprojb, CDIM * CDIM);
  k_bias_tab<<<48, 256, 0, stream>>>(Wg, bg, tab);

  k_qkv<<<dim3(129, 18), 128, 0, stream>>>(xb, wqkvb, Qb, Kb, Vt);
  k_attn<<<dim3(96, 17), 128, 0, stream>>>(Qb, Kb, Vt, tab, Ob);
  k_proj<<<dim3(129, 6), 128, 0, stream>>>(Ob, wprojb, bproj, out);
}
